// CustomLinear_36936718745709
// MI455X (gfx1250) — matrix operand order repaired and checked by reading, not compiled, not run on hardware
//
#include <hip/hip_runtime.h>

#define IN_F  1024
#define OUT_F 1024
#define NROWS 32768
#define BM 128
#define BN 128
#define KB 64
#define LDP 72   // padded LDS row stride (144B) -> conflict-friendly bank spread

typedef __attribute__((ext_vector_type(16))) __bf16 v16bf;
typedef __attribute__((ext_vector_type(8)))  __bf16 bf16x8;
typedef __attribute__((ext_vector_type(4)))  __bf16 bf16x4;
typedef __attribute__((ext_vector_type(8)))  float  v8f;
typedef __attribute__((ext_vector_type(4)))  float  f32x4;

// ---------------------------------------------------------------------------
// Kernel 1: fold the rewire delta into an effective weight matrix, stored as
// split bf16 (hi + lo): W_eff[r,s] = 0.25*(W[r,s]+sum_k W[r,c_k]),
// W_eff[r,c_k] = 0, everything else copied. out = x @ W_eff^T + bias exactly.
// ---------------------------------------------------------------------------
__global__ void build_weff(const float* __restrict__ weight,
                           const int*  __restrict__ rewire_rows,
                           const int*  __restrict__ rewire_src,
                           const int*  __restrict__ rewire_clones,
                           __bf16* __restrict__ wHi,
                           __bf16* __restrict__ wLo) {
    const int i  = blockIdx.x;
    const int r  = rewire_rows[i];
    const int s  = rewire_src[i];
    const int c0 = rewire_clones[i * 3 + 0];
    const int c1 = rewire_clones[i * 3 + 1];
    const int c2 = rewire_clones[i * 3 + 2];
    const float* wrow = weight + (long)r * IN_F;
    const float wsum = 0.25f * (wrow[s] + wrow[c0] + wrow[c1] + wrow[c2]);

    for (int j = threadIdx.x; j < IN_F; j += blockDim.x) {
        float w = wrow[j];
        if (j == s)                              w = wsum;
        else if (j == c0 || j == c1 || j == c2)  w = 0.0f;
        __bf16 h = (__bf16)w;
        __bf16 l = (__bf16)(w - (float)h);
        wHi[(long)r * IN_F + j] = h;
        wLo[(long)r * IN_F + j] = l;
    }
}

// ---------------------------------------------------------------------------
// Kernel 2: 128x128 block tile GEMM on v_wmma_f32_16x16x32_bf16.
//  - B (split-bf16 W_eff strip) staged to LDS via GLOBAL_LOAD_ASYNC_TO_LDS_B128
//    (ASYNCcnt), overlapping the A stage.
//  - A (f32 x strip) loaded to VGPRs, split to bf16 hi/lo, stored to LDS.
//  - Wave w computes rows [w*16, w*16+16) x all 128 block columns:
//    8 accumulator tiles, 3 WMMAs per tile per K-step (hi*hi + hi*lo + lo*hi).
// ---------------------------------------------------------------------------
__global__ __launch_bounds__(256, 2)
void gemm_rewire(const float*  __restrict__ x,
                 const __bf16* __restrict__ wHi,
                 const __bf16* __restrict__ wLo,
                 const float*  __restrict__ bias,
                 float*        __restrict__ out) {
    __shared__ __align__(16) __bf16 sAH[BM][LDP];
    __shared__ __align__(16) __bf16 sAL[BM][LDP];
    __shared__ __align__(16) __bf16 sBH[BN][LDP];
    __shared__ __align__(16) __bf16 sBL[BN][LDP];

    const int tid   = threadIdx.x;
    const int lane  = tid & 31;
    const int wave  = tid >> 5;
    const int m0    = blockIdx.x * BM;
    const int n0    = blockIdx.y * BN;
    const int lrow  = lane & 15;   // A: M, B: N, C: N within a 16x16 tile
    const int lhalf = lane >> 4;

    v8f acc[8] = {};

    for (int k0 = 0; k0 < IN_F; k0 += KB) {
        // ---- async-stage B: 2 x (128 x 64) bf16 = 2 x 1024 x 16B chunks ----
#pragma unroll
        for (int it = 0; it < 4; ++it) {
            const int c   = tid + it * 256;      // 0..1023
            const int row = c >> 3;              // 0..127 (W_eff row = out col)
            const int ko  = (c & 7) * 8;         // 0..56, 8 bf16 = 16B
            const __bf16* gH = wHi + (long)(n0 + row) * IN_F + k0 + ko;
            const __bf16* gL = wLo + (long)(n0 + row) * IN_F + k0 + ko;
            unsigned dH = (unsigned)(unsigned long long)&sBH[row][ko];
            unsigned dL = (unsigned)(unsigned long long)&sBL[row][ko];
            asm volatile("global_load_async_to_lds_b128 %0, %1, off"
                         :: "v"(dH), "v"(gH) : "memory");
            asm volatile("global_load_async_to_lds_b128 %0, %1, off"
                         :: "v"(dL), "v"(gL) : "memory");
        }

        // ---- stage A: 128 x 64 f32 -> split bf16 hi/lo in LDS ----
#pragma unroll
        for (int it = 0; it < 8; ++it) {
            const int c   = tid + it * 256;      // 0..2047 f32x4 chunks
            const int row = c >> 4;              // 0..127
            const int kq  = (c & 15) * 4;        // 0..60
            f32x4 v = *(const f32x4*)(x + (long)(m0 + row) * IN_F + k0 + kq);
            if (kq == 0 && k0 + KB < IN_F)       // hint next K-slab of x
                __builtin_prefetch(x + (long)(m0 + row) * IN_F + k0 + KB, 0, 3);
            bf16x4 h, l;
#pragma unroll
            for (int e = 0; e < 4; ++e) {
                h[e] = (__bf16)v[e];
                l[e] = (__bf16)(v[e] - (float)h[e]);
            }
            *(bf16x4*)&sAH[row][kq] = h;
            *(bf16x4*)&sAL[row][kq] = l;
        }

        // async B copies must land in LDS before anyone reads them
        asm volatile("s_wait_asynccnt 0x0" ::: "memory");
        __syncthreads();

        // ---- compute: 2 K-steps of 32, 8 N-tiles, 3 WMMAs each ----
#pragma unroll
        for (int ks = 0; ks < KB; ks += 32) {
            // A fragment (16-bit A 16x32): lane -> M = wave*16+lrow,
            // 16 consecutive K at lhalf*16
            const int am = wave * 16 + lrow;
            const int ca = ks + lhalf * 16;
            const int cb = ks + lhalf * 16 + 8;
            bf16x8 ah0 = *(const bf16x8*)&sAH[am][ca];
            bf16x8 ah1 = *(const bf16x8*)&sAH[am][cb];
            bf16x8 al0 = *(const bf16x8*)&sAL[am][ca];
            bf16x8 al1 = *(const bf16x8*)&sAL[am][cb];
            v16bf aH = __builtin_shufflevector(ah0, ah1, 0,1,2,3,4,5,6,7,8,9,10,11,12,13,14,15);
            v16bf aL = __builtin_shufflevector(al0, al1, 0,1,2,3,4,5,6,7,8,9,10,11,12,13,14,15);

#pragma unroll
            for (int nt = 0; nt < 8; ++nt) {
                // B fragment (16-bit B 32x16): lane -> N = nt*16+lrow,
                // 16 consecutive K at lhalf*16
                const int bn = nt * 16 + lrow;
                const int kc = ks + lhalf * 16;
                bf16x8 bh0 = *(const bf16x8*)&sBH[bn][kc];
                bf16x8 bh1 = *(const bf16x8*)&sBH[bn][kc + 8];
                bf16x8 bl0 = *(const bf16x8*)&sBL[bn][kc];
                bf16x8 bl1 = *(const bf16x8*)&sBL[bn][kc + 8];
                v16bf bH = __builtin_shufflevector(bh0, bh1, 0,1,2,3,4,5,6,7,8,9,10,11,12,13,14,15);
                v16bf bL = __builtin_shufflevector(bl0, bl1, 0,1,2,3,4,5,6,7,8,9,10,11,12,13,14,15);

                acc[nt] = __builtin_amdgcn_wmma_f32_16x16x32_bf16(
                              false, aH, false, bH, (short)0, acc[nt], false, false);
                acc[nt] = __builtin_amdgcn_wmma_f32_16x16x32_bf16(
                              false, aH, false, bL, (short)0, acc[nt], false, false);
                acc[nt] = __builtin_amdgcn_wmma_f32_16x16x32_bf16(
                              false, aL, false, bH, (short)0, acc[nt], false, false);
            }
        }
        __syncthreads();
    }

    // ---- epilogue: C layout VGPR v -> M = v + 8*lhalf, N = lrow ----
#pragma unroll
    for (int nt = 0; nt < 8; ++nt) {
        const int   n  = n0 + nt * 16 + lrow;
        const float bv = bias[n];
#pragma unroll
        for (int vv = 0; vv < 8; ++vv) {
            const int row = m0 + wave * 16 + vv + lhalf * 8;
            out[(long)row * OUT_F + n] = acc[nt][vv] + bv;
        }
    }
}

extern "C" void kernel_launch(void* const* d_in, const int* in_sizes, int n_in,
                              void* d_out, int out_size, void* d_ws, size_t ws_size,
                              hipStream_t stream) {
    const float* x             = (const float*)d_in[0];
    const float* weight        = (const float*)d_in[1];
    const float* bias          = (const float*)d_in[2];
    const int*   rewire_rows   = (const int*)d_in[3];
    const int*   rewire_src    = (const int*)d_in[4];
    const int*   rewire_clones = (const int*)d_in[5];
    float*       out           = (float*)d_out;

    // workspace: split-bf16 effective weights (2 MB + 2 MB)
    __bf16* wHi = (__bf16*)d_ws;
    __bf16* wLo = wHi + (size_t)OUT_F * IN_F;

    build_weff<<<OUT_F, 256, 0, stream>>>(weight, rewire_rows, rewire_src,
                                          rewire_clones, wHi, wLo);
    dim3 grid(NROWS / BM, OUT_F / BN);
    gemm_rewire<<<grid, 256, 0, stream>>>(x, wHi, wLo, bias, out);
}
